// CNFModel_27831388078299
// MI455X (gfx1250) — matrix operand order repaired and checked by reading, not compiled, not run on hardware
//
#include <hip/hip_runtime.h>
#include <hip/hip_bf16.h>

// ---------------------------------------------------------------------------
// CNF log-density: fused RK45 (Dormand-Prince) x 4 steps, each step = 6 JVP
// evaluations of a 64->256->256->256->64 tanh MLP. All matmuls via
// v_wmma_f32_16x16x32_f16 (CDNA5 / gfx1250, wave32).
// ---------------------------------------------------------------------------

typedef _Float16 half_t;
typedef __attribute__((ext_vector_type(16))) _Float16 v16h;
typedef __attribute__((ext_vector_type(8)))  _Float16 v8h;
typedef __attribute__((ext_vector_type(8)))  float    v8f;

#define DIM   64
#define HID   256
#define BATCH 32768
#define NSTEP 4
#define STR   264   // LDS row stride in halves (multiple of 8 -> 16B aligned)

#define LDS_FENCE() asm volatile("s_wait_dscnt 0" ::: "memory")

// tanh via v_exp_f32 + v_rcp_f32 (no IEEE-divide expansion):
// tanh(x) = 1 - 2/(1 + e^{2x})
__device__ __forceinline__ float fast_tanh(float x) {
    float e = __expf(2.0f * x);
    float r = __builtin_amdgcn_rcpf(1.0f + e);
    return 1.0f - 2.0f * r;
}

// A-fragment (16x32 f16) from a row-major [16][STR] LDS tile.
// Lane L: M = L&15 ; halves 0..7 = K = kbase + (L>>4)*8 + j,
//                     halves 8..15 = K = kbase + 16 + (L>>4)*8 + j.
__device__ __forceinline__ v16h ldsAfrag(const half_t* buf, int lo, int hi, int kbase) {
    const half_t* p = buf + lo * STR + kbase + hi * 8;
    v8h a0 = *(const v8h*)(p);
    v8h a1 = *(const v8h*)(p + 16);
    v16h r;
#pragma unroll
    for (int j = 0; j < 8; ++j) { r[j] = a0[j]; r[8 + j] = a1[j]; }
    return r;
}

// B-fragment (32x16 f16) straight from row-major W (NOUT x K):
// lane L: N = nbase + (L&15); halves 0..7 = K = kbase + (L>>4)*8 + j,
//                             halves 8..15 = K = kbase + 16 + (L>>4)*8 + j.
__device__ __forceinline__ v16h glBfrag(const half_t* W, int K, int nbase,
                                        int lo, int hi, int kbase) {
    const half_t* p = W + (size_t)(nbase + lo) * K + kbase + hi * 8;
    v8h b0 = *(const v8h*)(p);
    v8h b1 = *(const v8h*)(p + 16);
    v16h r;
#pragma unroll
    for (int j = 0; j < 8; ++j) { r[j] = b0[j]; r[8 + j] = b1[j]; }
    return r;
}

__device__ __forceinline__ v8f wmma_f16(v16h a, v16h b, v8f c) {
    return __builtin_amdgcn_wmma_f32_16x16x32_f16(false, a, false, b, (short)0, c,
                                                  false, false);
}

// One hidden layer: primal + tangent share each B fragment. Writes tanh(h)
// and (1-tanh^2)*t back to LDS staging (f16) for the next layer's A frags.
template<int K, bool FRAG_T>
__device__ __forceinline__ void layerFwd(const half_t* inH, const half_t* inT,
                                         const v16h* tFrags,
                                         const half_t* W, const float* bias,
                                         half_t* outH, half_t* outT,
                                         int lo, int hi) {
#pragma unroll 1
    for (int n2 = 0; n2 < HID / 16; ++n2) {
        float bv = bias[n2 * 16 + lo];
        v8f accP, accT;
#pragma unroll
        for (int i = 0; i < 8; ++i) { accP[i] = bv; accT[i] = 0.0f; }
#pragma unroll
        for (int kf = 0; kf < K / 32; ++kf) {
            v16h bF = glBfrag(W, K, n2 * 16, lo, hi, kf * 32);
            v16h aH = ldsAfrag(inH, lo, hi, kf * 32);
            v16h aT;
            if constexpr (FRAG_T) aT = tFrags[kf];
            else                  aT = ldsAfrag(inT, lo, hi, kf * 32);
            accP = wmma_f16(aH, bF, accP);
            accT = wmma_f16(aT, bF, accT);
        }
        const int r0 = 8 * hi;
#pragma unroll
        for (int i = 0; i < 8; ++i) {
            float th = fast_tanh(accP[i]);
            float tv = accT[i] * (1.0f - th * th);
            outH[(r0 + i) * STR + n2 * 16 + lo] = (half_t)th;
            outT[(r0 + i) * STR + n2 * 16 + lo] = (half_t)tv;
        }
    }
}

__global__ void __launch_bounds__(128, 1)
cnf_rk45_kernel(const float* __restrict__ x, const float* __restrict__ eps,
                const float* __restrict__ b1, const float* __restrict__ b2,
                const float* __restrict__ b3, const float* __restrict__ b4,
                const half_t* __restrict__ ws, float* __restrict__ out) {
    __shared__ half_t smem[4][4][16 * STR];   // [wave][H0,H1,T0,T1]

    const int tid  = threadIdx.x;
    const int lane = tid & 31;
    const int wv   = tid >> 5;
    const int lo   = lane & 15;
    const int hi   = lane >> 4;
    const int rowBase = blockIdx.x * 64 + wv * 16;

    half_t* H0 = smem[wv][0];
    half_t* H1 = smem[wv][1];
    half_t* T0 = smem[wv][2];
    half_t* T1 = smem[wv][3];

    const half_t* W1h = ws;
    const half_t* W2h = ws + 16384;
    const half_t* W3h = ws + 81920;
    const half_t* W4h = ws + 147456;

    // ---- ODE state: z in WMMA C-layout (tile n covers cols 16n..16n+15;
    // vgpr i -> row i + 8*hi; lane&15 -> col). logp replicated per 16-lane half.
    v8f yz[4];
    float yp[8];
#pragma unroll
    for (int n = 0; n < 4; ++n)
#pragma unroll
        for (int i = 0; i < 8; ++i)
            yz[n][i] = x[(size_t)(rowBase + 8 * hi + i) * DIM + n * 16 + lo];
#pragma unroll
    for (int i = 0; i < 8; ++i) yp[i] = 0.0f;

    // eps in C-layout f32 (for the divergence dot product)
    v8f ce[4];
#pragma unroll
    for (int n = 0; n < 4; ++n)
#pragma unroll
        for (int i = 0; i < 8; ++i)
            ce[n][i] = eps[(size_t)(rowBase + 8 * hi + i) * DIM + n * 16 + lo];

    // eps as f16 A-fragments for layer-1 tangent (constant over all evals)
    v16h epsFrag[2];
#pragma unroll
    for (int kf = 0; kf < 2; ++kf) {
        const float* er = eps + (size_t)(rowBase + lo) * DIM + kf * 32 + hi * 8;
#pragma unroll
        for (int j = 0; j < 8; ++j) {
            epsFrag[kf][j]     = (half_t)er[j];
            epsFrag[kf][8 + j] = (half_t)er[16 + j];
        }
    }

    // ---- one odefunc evaluation: zin -> (kz, kd) ----
    auto evalF = [&](const v8f (&zin)[4], v8f (&kz)[4], float (&kd)[8]) {
        // stage z (f16, row-major) into H0
#pragma unroll
        for (int n = 0; n < 4; ++n)
#pragma unroll
            for (int i = 0; i < 8; ++i)
                H0[(8 * hi + i) * STR + n * 16 + lo] = (half_t)zin[n][i];
        LDS_FENCE();
        layerFwd<DIM, true >(H0, (const half_t*)nullptr, epsFrag, W1h, b1, H1, T1, lo, hi);
        LDS_FENCE();
        layerFwd<HID, false>(H1, T1, (const v16h*)nullptr, W2h, b2, H0, T0, lo, hi);
        LDS_FENCE();
        layerFwd<HID, false>(H0, T0, (const v16h*)nullptr, W3h, b3, H1, T1, lo, hi);
        LDS_FENCE();
        // output layer: dz (no tanh) + divergence accumulation
        float dvp[8];
#pragma unroll
        for (int i = 0; i < 8; ++i) dvp[i] = 0.0f;
#pragma unroll
        for (int n2 = 0; n2 < 4; ++n2) {
            float bv = b4[n2 * 16 + lo];
            v8f aP, aT;
#pragma unroll
            for (int i = 0; i < 8; ++i) { aP[i] = bv; aT[i] = 0.0f; }
#pragma unroll
            for (int kf = 0; kf < HID / 32; ++kf) {
                v16h bF  = glBfrag(W4h, HID, n2 * 16, lo, hi, kf * 32);
                v16h aH  = ldsAfrag(H1, lo, hi, kf * 32);
                v16h aTt = ldsAfrag(T1, lo, hi, kf * 32);
                aP = wmma_f16(aH,  bF, aP);
                aT = wmma_f16(aTt, bF, aT);
            }
            kz[n2] = aP;
#pragma unroll
            for (int i = 0; i < 8; ++i) dvp[i] += aT[i] * ce[n2][i];
        }
#pragma unroll
        for (int i = 0; i < 8; ++i) {
            float d = dvp[i];
            d += __shfl_xor(d, 1);
            d += __shfl_xor(d, 2);
            d += __shfl_xor(d, 4);
            d += __shfl_xor(d, 8);
            kd[i] = -d;
        }
    };

    // ---- Dormand-Prince RK45 ----
    const float h = 1.0f / NSTEP;
    const float B0 = 35.0f/384.0f, B2 = 500.0f/1113.0f, B3 = 125.0f/192.0f,
                B4 = -2187.0f/6784.0f, B5 = 11.0f/84.0f;
    v8f k0z[4], k1z[4], k2z[4], k3z[4], k4z[4], tz[4];
    float kd[8];

#pragma unroll 1
    for (int s = 0; s < NSTEP; ++s) {
        // stage 0
        evalF(yz, k0z, kd);
#pragma unroll
        for (int i = 0; i < 8; ++i) yp[i] += (h * B0) * kd[i];
#pragma unroll
        for (int n = 0; n < 4; ++n)
            tz[n] = yz[n] + (h * 0.2f) * k0z[n];
        // stage 1 (b1 == 0: no yp contribution)
        evalF(tz, k1z, kd);
#pragma unroll
        for (int n = 0; n < 4; ++n)
            tz[n] = yz[n] + (h * (3.0f/40.0f)) * k0z[n] + (h * (9.0f/40.0f)) * k1z[n];
        // stage 2
        evalF(tz, k2z, kd);
#pragma unroll
        for (int i = 0; i < 8; ++i) yp[i] += (h * B2) * kd[i];
#pragma unroll
        for (int n = 0; n < 4; ++n)
            tz[n] = yz[n] + (h * (44.0f/45.0f)) * k0z[n] + (h * (-56.0f/15.0f)) * k1z[n]
                          + (h * (32.0f/9.0f))  * k2z[n];
        // stage 3
        evalF(tz, k3z, kd);
#pragma unroll
        for (int i = 0; i < 8; ++i) yp[i] += (h * B3) * kd[i];
#pragma unroll
        for (int n = 0; n < 4; ++n)
            tz[n] = yz[n] + (h * (19372.0f/6561.0f))  * k0z[n]
                          + (h * (-25360.0f/2187.0f)) * k1z[n]
                          + (h * (64448.0f/6561.0f))  * k2z[n]
                          + (h * (-212.0f/729.0f))    * k3z[n];
        // stage 4
        evalF(tz, k4z, kd);
#pragma unroll
        for (int i = 0; i < 8; ++i) yp[i] += (h * B4) * kd[i];
#pragma unroll
        for (int n = 0; n < 4; ++n)
            tz[n] = yz[n] + (h * (9017.0f/3168.0f))   * k0z[n]
                          + (h * (-355.0f/33.0f))     * k1z[n]
                          + (h * (46732.0f/5247.0f))  * k2z[n]
                          + (h * (49.0f/176.0f))      * k3z[n]
                          + (h * (-5103.0f/18656.0f)) * k4z[n];
        // fold the b-weighted partial update now: k0..k4 become dead,
        // so stage 5 can reuse k1z's registers.
#pragma unroll
        for (int n = 0; n < 4; ++n)
            yz[n] = yz[n] + h * (B0 * k0z[n] + B2 * k2z[n] + B3 * k3z[n] + B4 * k4z[n]);
        // stage 5
        evalF(tz, k1z, kd);
#pragma unroll
        for (int i = 0; i < 8; ++i) yp[i] += (h * B5) * kd[i];
#pragma unroll
        for (int n = 0; n < 4; ++n)
            yz[n] = yz[n] + (h * B5) * k1z[n];
    }

    // ---- log p(x) = sum(-0.5 z^2 - 0.5 log 2pi) - logpT ----
    const float LOG_2PI = 1.8378770664093453f;
#pragma unroll
    for (int i = 0; i < 8; ++i) {
        float acc = 0.0f;
#pragma unroll
        for (int n = 0; n < 4; ++n) acc += yz[n][i] * yz[n][i];
        acc *= -0.5f;
        acc += __shfl_xor(acc, 1);
        acc += __shfl_xor(acc, 2);
        acc += __shfl_xor(acc, 4);
        acc += __shfl_xor(acc, 8);
        acc -= 0.5f * LOG_2PI * (float)DIM;
        if (lo == 0)
            out[rowBase + 8 * hi + i] = acc - yp[i];
    }
}

// f32 -> f16 weight conversion into workspace
__global__ void cvt_weights_kernel(const float* __restrict__ W1,
                                   const float* __restrict__ W2,
                                   const float* __restrict__ W3,
                                   const float* __restrict__ W4,
                                   half_t* __restrict__ ws) {
    int i = blockIdx.x * blockDim.x + threadIdx.x;
    if (i < 16384) ws[i]          = (half_t)W1[i];
    if (i < 65536) ws[16384 + i]  = (half_t)W2[i];
    if (i < 65536) ws[81920 + i]  = (half_t)W3[i];
    if (i < 16384) ws[147456 + i] = (half_t)W4[i];
}

extern "C" void kernel_launch(void* const* d_in, const int* in_sizes, int n_in,
                              void* d_out, int out_size, void* d_ws, size_t ws_size,
                              hipStream_t stream) {
    const float* x   = (const float*)d_in[0];
    const float* eps = (const float*)d_in[1];
    const float* W1  = (const float*)d_in[2];
    const float* b1  = (const float*)d_in[3];
    const float* W2  = (const float*)d_in[4];
    const float* b2  = (const float*)d_in[5];
    const float* W3  = (const float*)d_in[6];
    const float* b3  = (const float*)d_in[7];
    const float* W4  = (const float*)d_in[8];
    const float* b4  = (const float*)d_in[9];
    float* out   = (float*)d_out;
    half_t* wsh  = (half_t*)d_ws;

    cvt_weights_kernel<<<256, 256, 0, stream>>>(W1, W2, W3, W4, wsh);

    // 16 rows per wave, 4 waves per block -> 64 rows per block
    cnf_rk45_kernel<<<BATCH / 64, 128, 0, stream>>>(x, eps, b1, b2, b3, b4, wsh, out);
}
